// AttentionHead_Hybrid2_8315056685233
// MI455X (gfx1250) — matrix operand order repaired and checked by reading, not compiled, not run on hardware
//
#include <hip/hip_runtime.h>
#include <hip/hip_fp16.h>

typedef __attribute__((ext_vector_type(16))) _Float16 v16h;
typedef __attribute__((ext_vector_type(8)))  float    v8f;

#define B_ 16
#define S_ 2048
#define D_ 64

// Workspace layout:
//   Vt : _Float16 [B][D][S]  (V transposed -> contiguous K-dim for WMMA B frags)  4 MiB
//   Qf : float    [B][S]                                                         128 KiB
//   Kf : float    [B][S]                                                         128 KiB

__global__ __launch_bounds__(64)
void proj_qkv_kernel(const float* __restrict__ x,  const float* __restrict__ Wv,
                     const float* __restrict__ Wq, const float* __restrict__ Wk,
                     _Float16* __restrict__ Vt, float* __restrict__ Qf,
                     float* __restrict__ Kf)
{
    __shared__ float xs[D_];
    __shared__ float qred[D_];
    __shared__ float kred[D_];
    const int t  = threadIdx.x;          // output channel e = t
    const int bs = blockIdx.x;           // token index b*S + s
    const int b  = bs >> 11;
    const int s  = bs & (S_ - 1);

    const float xv = x[(size_t)bs * D_ + t];
    xs[t]   = xv;
    qred[t] = xv * Wq[t];
    kred[t] = xv * Wk[t];
    __syncthreads();

    float acc = 0.f;
#pragma unroll 8
    for (int d = 0; d < D_; ++d)
        acc = fmaf(xs[d], Wv[d * D_ + t], acc);
    // V[b][s][t] stored transposed: Vt[b][t][s]
    Vt[((size_t)(b * D_ + t)) * S_ + s] = (_Float16)acc;

    if (t == 0) {
        float q = 0.f, k = 0.f;
        for (int i = 0; i < D_; ++i) { q += qred[i]; k += kred[i]; }
        Qf[b * S_ + s] = q;
        Kf[b * S_ + s] = k;
    }
}

__global__ __launch_bounds__(128)
void attn_wmma_kernel(const _Float16* __restrict__ Vt, const float* __restrict__ Qf,
                      const float* __restrict__ Kf, float* __restrict__ out)
{
    __shared__ float oacc[16 * D_];   // 16 rows x 64 cols f32 partial-sum tile
    __shared__ float sred[16];        // softmax denominators

    const int tid  = threadIdx.x;
    const int lane = tid & 31;
    const int wave = tid >> 5;        // 4 waves; each owns its own j-chunk per step
    const int i0   = blockIdx.x * 16; // row tile
    const int b    = blockIdx.y;

    for (int e = tid; e < 16 * D_; e += 128) oacc[e] = 0.f;
    if (tid < 16) sred[tid] = 0.f;
    __syncthreads();

    const int  m     = lane & 15;         // matrix row this lane owns (A; n-col for B/C)
    const bool hi    = lane >= 16;
    const int  koffA = hi ? 16 : 0;       // A fragment: half-wave k offset
    const int  koffB = hi ? 16 : 0;       // B fragment: half-wave k offset
    const int  jlane = (wave << 5);       // this wave's 32-wide chunk within a 128 step

    const float q = Qf[b * S_ + i0 + m];
    const float* __restrict__ Kb = Kf + b * S_;
    const _Float16* __restrict__ vp0 =
        Vt + (size_t)b * D_ * S_ + (size_t)m * S_;   // column d = 16*slice + m

    v8f c0 = {}, c1 = {}, c2 = {}, c3 = {};

    // ---- software pipeline: preload tile 0 ----
    float4 ka, kb, kc, kd;
    v16h   b0, b1, b2, b3;
    {
        const float* kp = Kb + jlane + koffA;
        ka = *(const float4*)(kp);       kb = *(const float4*)(kp + 4);
        kc = *(const float4*)(kp + 8);   kd = *(const float4*)(kp + 12);
        const _Float16* vp = vp0 + jlane + koffB;
        b0 = *(const v16h*)(vp);
        b1 = *(const v16h*)(vp + 16 * S_);
        b2 = *(const v16h*)(vp + 32 * S_);
        b3 = *(const v16h*)(vp + 48 * S_);
    }

    for (int jt = 0; jt < S_; jt += 128) {
        // ---- prefetch next tile first; wraps on last iter (harmless L2 re-read),
        //      keeps control flow branch-free and EXEC all-ones for WMMA ----
        const int jn = (jt + 128) & (S_ - 1);
        float4 na, nb_, nc, nd;
        v16h   n0, n1, n2, n3;
        {
            const float* kp = Kb + jn + jlane + koffA;
            na  = *(const float4*)(kp);       nb_ = *(const float4*)(kp + 4);
            nc  = *(const float4*)(kp + 8);   nd  = *(const float4*)(kp + 12);
            const _Float16* vp = vp0 + jn + jlane + koffB;
            n0 = *(const v16h*)(vp);
            n1 = *(const v16h*)(vp + 16 * S_);
            n2 = *(const v16h*)(vp + 32 * S_);
            n3 = *(const v16h*)(vp + 48 * S_);
        }

        // ---- A fragment: 16 exp weights per lane (hides prefetch latency) ----
        const float kv[16] = { ka.x, ka.y, ka.z, ka.w,  kb.x, kb.y, kb.z, kb.w,
                               kc.x, kc.y, kc.z, kc.w,  kd.x, kd.y, kd.z, kd.w };
        v16h a;
        float rs = 0.f;
#pragma unroll
        for (int e = 0; e < 16; ++e) {
            const float dlt = q - kv[e];
            const float w   = __expf(dlt * dlt * -0.125f);  // in (0,1]: stable, no max pass
            rs  += w;
            a[e] = (_Float16)w;
        }

        // ---- 4 d-slices: out[16x64] partial accumulate ----
        c0 = __builtin_amdgcn_wmma_f32_16x16x32_f16(false, a, false, b0, (short)0, c0, false, false);
        c1 = __builtin_amdgcn_wmma_f32_16x16x32_f16(false, a, false, b1, (short)0, c1, false, false);
        c2 = __builtin_amdgcn_wmma_f32_16x16x32_f16(false, a, false, b2, (short)0, c2, false, false);
        c3 = __builtin_amdgcn_wmma_f32_16x16x32_f16(false, a, false, b3, (short)0, c3, false, false);

        atomicAdd(&sred[m], rs);          // ds_add_f32 (no return)

        // ---- rotate pipeline registers ----
        ka = na; kb = nb_; kc = nc; kd = nd;
        b0 = n0; b1 = n1; b2 = n2; b3 = n3;
    }

    // Cross-wave reduction: C/D layout -> lane holds (mrow = r + hi*8, n = lane&15)
#pragma unroll
    for (int r = 0; r < 8; ++r) {
        const int mc = r + (hi ? 8 : 0);
        atomicAdd(&oacc[mc * D_ + m     ], c0[r]);
        atomicAdd(&oacc[mc * D_ + m + 16], c1[r]);
        atomicAdd(&oacc[mc * D_ + m + 32], c2[r]);
        atomicAdd(&oacc[mc * D_ + m + 48], c3[r]);
    }
    __syncthreads();

    for (int e = tid; e < 16 * D_; e += 128) {
        const int mm = e >> 6, dd = e & 63;
        out[((size_t)(b * S_ + i0 + mm)) * D_ + dd] = oacc[e] / sred[mm];
    }
}

extern "C" void kernel_launch(void* const* d_in, const int* in_sizes, int n_in,
                              void* d_out, int out_size, void* d_ws, size_t ws_size,
                              hipStream_t stream)
{
    (void)in_sizes; (void)n_in; (void)out_size; (void)ws_size;
    const float* x  = (const float*)d_in[0];
    const float* Wv = (const float*)d_in[1];
    const float* Wq = (const float*)d_in[2];
    const float* Wk = (const float*)d_in[3];

    char* ws = (char*)d_ws;
    _Float16* Vt = (_Float16*)ws;                                   // 4 MiB
    float*    Qf = (float*)(ws + (size_t)B_ * D_ * S_ * sizeof(_Float16));
    float*    Kf = Qf + (size_t)B_ * S_;
    float*    out = (float*)d_out;

    proj_qkv_kernel<<<dim3(B_ * S_), dim3(64), 0, stream>>>(x, Wv, Wq, Wk, Vt, Qf, Kf);
    attn_wmma_kernel<<<dim3(S_ / 16, B_), dim3(128), 0, stream>>>(Vt, Qf, Kf, out);
}
